// SpatialGatingUnit_59330678226987
// MI455X (gfx1250) — matrix operand order repaired and checked by reading, not compiled, not run on hardware
//
#include <hip/hip_runtime.h>
#include <hip/hip_bf16.h>

typedef __bf16 bf16;
typedef __attribute__((ext_vector_type(16))) __bf16 v16bf;
typedef __attribute__((ext_vector_type(8)))  float  v8f;
typedef __attribute__((ext_vector_type(4)))  unsigned int uint4v;
typedef __attribute__((ext_vector_type(8)))  int int8v;
typedef __attribute__((ext_vector_type(4)))  int int4v;

#define TT   1024   // T (sequence length)
#define HD   512    // H (hidden)
#define BSZ  32     // batch
#define BM   128
#define BN   64
#define BK   32
#define LN_EPS 1e-5f

#if defined(__gfx1250__) && __has_builtin(__builtin_amdgcn_tensor_load_to_lds) && \
    __has_builtin(__builtin_amdgcn_s_wait_tensorcnt)
#define HAVE_TDM 1
#else
#define HAVE_TDM 0
#endif

union ABFrag { v16bf v; unsigned int u[8]; };
union CDFrag { v8f v; float f[8]; };
union PairU  { unsigned int u; bf16 h[2]; };

// ---------------------------------------------------------------------------
// Kernel 0: W (fp32, [T,T]) -> bf16 with lower-triangular mask applied.
// ---------------------------------------------------------------------------
__global__ __launch_bounds__(256) void sgu_mask_cast_w(const float* __restrict__ W,
                                                       bf16* __restrict__ Wbf) {
    int i = blockIdx.x * 256 + threadIdx.x;      // 0 .. T*T-1
    int r = i >> 10;                             // row
    int c = i & (TT - 1);                        // col
    float w = (c <= r) ? W[i] : 0.0f;
    Wbf[i] = (bf16)w;
}

// ---------------------------------------------------------------------------
// Kernel 1: LayerNorm over H for v = x[..., H:2H]; write bf16 v_hat.
// ---------------------------------------------------------------------------
__global__ __launch_bounds__(256) void sgu_layernorm(const float* __restrict__ x,
                                                     const float* __restrict__ gamma,
                                                     const float* __restrict__ beta,
                                                     bf16* __restrict__ Vn) {
    const int row = blockIdx.x;                           // 0 .. B*T-1
    const float* v = x + (size_t)row * (2 * HD) + HD;
    const int tid  = threadIdx.x;

    float a = v[tid];
    float c = v[tid + 256];
    float s  = a + c;
    float s2 = a * a + c * c;

    #pragma unroll
    for (int off = 16; off > 0; off >>= 1) {
        s  += __shfl_xor(s,  off, 32);
        s2 += __shfl_xor(s2, off, 32);
    }

    __shared__ float ps[8], ps2[8];
    __shared__ float mu_s, ri_s;
    const int w = tid >> 5, l = tid & 31;
    if (l == 0) { ps[w] = s; ps2[w] = s2; }
    __syncthreads();
    if (tid == 0) {
        float ts = 0.f, ts2 = 0.f;
        #pragma unroll
        for (int i = 0; i < 8; ++i) { ts += ps[i]; ts2 += ps2[i]; }
        float mu  = ts * (1.0f / HD);
        float var = ts2 * (1.0f / HD) - mu * mu;
        mu_s = mu;
        ri_s = __frsqrt_rn(var + LN_EPS);
    }
    __syncthreads();
    const float mu = mu_s, ri = ri_s;

    bf16* o = Vn + (size_t)row * HD;
    o[tid]       = (bf16)(((a - mu) * ri) * gamma[tid]       + beta[tid]);
    o[tid + 256] = (bf16)(((c - mu) * ri) * gamma[tid + 256] + beta[tid + 256]);
}

// ---------------------------------------------------------------------------
// TDM helper: DMA a 128x32 bf16 tile (row stride TT) into LDS laid out as
// [128][BK+2] (pad_interval = 16 dwords, pad_amount = 1 dword).
// D# packing per CDNA5 ISA sec. 8.3 / 8.4.
// ---------------------------------------------------------------------------
#if HAVE_TDM
__device__ __forceinline__ unsigned lds_addr_of(void* p) {
    return (unsigned)(unsigned long long)(__attribute__((address_space(3))) void*)p;
}

__device__ __forceinline__ void tdm_load_w_tile(const bf16* gsrc, unsigned lds_byte) {
    unsigned long long ga = (unsigned long long)gsrc;
    uint4v g0;
    g0.x = 1u;                                      // count=1 (user mode)
    g0.y = lds_byte;                                // lds_addr
    g0.z = (unsigned)(ga & 0xFFFFFFFFu);            // global_addr[31:0]
    g0.w = (unsigned)((ga >> 32) & 0x01FFFFFFu)     // global_addr[56:32]
         | (2u << 30);                              // type = 2 ("image")
    int8v g1;
    g1[0] = (1 << 16)                               // data_size = 1 -> 2 bytes
          | (1 << 20)                               // pad_enable
          | (3 << 22);                              // pad_interval code 3 = 16 dwords
                                                    // pad_amount = 0 -> 1 dword (2 bf16)
    g1[1] = (int)((unsigned)TT << 16);              // tensor_dim0[15:0] = 1024
    g1[2] = (int)((unsigned)TT << 16);              // tensor_dim1[15:0] = 1024
    g1[3] = (int)(32u << 16);                       // tile_dim0 = 32 (K)
    g1[4] = 128;                                    // tile_dim1 = 128 (rows), tile_dim2=0
    g1[5] = TT;                                     // tensor_dim0_stride[31:0] = 1024
    g1[6] = 0;                                      // stride hi / dim1_stride lo
    g1[7] = 0;                                      // dim1_stride hi
    int4v gz = {0, 0, 0, 0};
#if __clang_major__ >= 23
    int8v gz8 = {0, 0, 0, 0, 0, 0, 0, 0};
    __builtin_amdgcn_tensor_load_to_lds(g0, g1, gz, gz, gz8, 0);
#else
    __builtin_amdgcn_tensor_load_to_lds(g0, g1, gz, gz, 0);
#endif
}
#endif

// ---------------------------------------------------------------------------
// Kernel 2: causal GEMM via v_wmma_f32_16x16x32_bf16 + fused bias + u-gate.
//   out[b,m,n] = u[b,m,n] * ( sum_k Wtril[m,k] * vhat[b,k,n] + bias[m] )
// Grid: (H/BN, T/BM, B); block 256 = 8 waves (4m x 2n); wave = 32x32 block.
// W tile staged by TDM (tensor_load_to_lds); V tile staged transposed by VALU.
// ---------------------------------------------------------------------------
__global__ __launch_bounds__(256) void sgu_gemm_wmma(const bf16*  __restrict__ Wbf,
                                                     const bf16*  __restrict__ Vn,
                                                     const float* __restrict__ x,
                                                     const float* __restrict__ bias,
                                                     float*       __restrict__ out) {
    const int nb   = blockIdx.x;          // n-tile: 0..H/BN-1
    const int mb   = blockIdx.y;          // m-tile: 0..T/BM-1
    const int b    = blockIdx.z;          // batch
    const int tid  = threadIdx.x;
    const int wave = tid >> 5;
    const int lane = tid & 31;
    const int wm   = wave & 3;            // wave row 0..3
    const int wn   = wave >> 2;           // wave col 0..1
    const int m_base = mb * BM;
    const int n_base = nb * BN;

    // LDS tiles (padded rows: +2 bf16 -> row stride 17 dwords, conflict-free)
    __shared__ bf16 Wt[BM][BK + 2];       // A tile, (m, k) row-major
    __shared__ bf16 Vt[BN][BK + 2];       // B tile, TRANSPOSED to (n, k)

    CDFrag acc[2][2];
    #pragma unroll
    for (int i = 0; i < 2; ++i)
        #pragma unroll
        for (int j = 0; j < 2; ++j)
            #pragma unroll
            for (int r = 0; r < 8; ++r) acc[i][j].f[r] = 0.0f;

    const bf16* Vb = Vn + (size_t)b * TT * HD;
    const int hlf = lane >> 4;            // lane half (0/1)
    const int ml  = lane & 15;
    const int k_end = m_base + BM;        // triangular: only K <= diag block

#if HAVE_TDM
    const unsigned wt_lds = lds_addr_of(&Wt[0][0]);
#endif

    for (int k0 = 0; k0 < k_end; k0 += BK) {
#if HAVE_TDM
        // ---- stage A tile via Tensor Data Mover (one wave issues DMA) ----
        if (tid < 32) {
            tdm_load_w_tile(Wbf + (size_t)m_base * TT + k0, wt_lds);
        }
#else
        // ---- stage A tile manually: 128x32 bf16 = 2048 dwords ----
        #pragma unroll
        for (int t = 0; t < 8; ++t) {
            int idx = tid + t * 256;            // dword index
            int r   = idx >> 4;                 // row (16 dwords/row)
            int kc  = (idx & 15) << 1;          // even k
            unsigned int d = *(const unsigned int*)(Wbf + (size_t)(m_base + r) * TT + k0 + kc);
            *(unsigned int*)&Wt[r][kc] = d;
        }
#endif
        // ---- stage B tile transposed: (32 k) x (64 n), 4 dwords/thread ----
        #pragma unroll
        for (int t = 0; t < 4; ++t) {
            int idx = tid + t * 256;            // dword index (1024 total)
            int k   = idx >> 5;                 // 0..31
            int n   = (idx & 31) << 1;          // even n
            PairU p;
            p.u = *(const unsigned int*)(Vb + (size_t)(k0 + k) * HD + n_base + n);
            Vt[n][k]     = p.h[0];
            Vt[n + 1][k] = p.h[1];
        }
#if HAVE_TDM
        if (tid < 32) __builtin_amdgcn_s_wait_tensorcnt(0);
#endif
        __syncthreads();

        // ---- build fragments per ISA 7.12.2 16-bit layouts ----
        ABFrag afrag[2], bfrag[2];
        #pragma unroll
        for (int ti = 0; ti < 2; ++ti) {
            int m = wm * 32 + ti * 16 + ml;
            #pragma unroll
            for (int p = 0; p < 8; ++p) {
                // A 16x32: VGPR p<4 -> K=2p(+8*half); p>=4 -> K=16+2(p-4)(+8*half)
                int k = (p < 4 ? 2 * p : 8 + 2 * p) + 8 * hlf;
                afrag[ti].u[p] = *(const unsigned int*)&Wt[m][k];
            }
        }
        #pragma unroll
        for (int tj = 0; tj < 2; ++tj) {
            int n = wn * 32 + tj * 16 + ml;
            #pragma unroll
            for (int p = 0; p < 8; ++p) {
                // B 32x16: lane=N, VGPR p<4 -> K=2p(+8*half); p>=4 -> K=16+2(p-4)(+8*half)
                int k = (p < 4 ? 2 * p : 8 + 2 * p) + 8 * hlf;
                bfrag[tj].u[p] = *(const unsigned int*)&Vt[n][k];
            }
        }

        #pragma unroll
        for (int ti = 0; ti < 2; ++ti)
            #pragma unroll
            for (int tj = 0; tj < 2; ++tj)
                acc[ti][tj].v = __builtin_amdgcn_wmma_f32_16x16x32_bf16(
                    /*neg_a=*/false, afrag[ti].v,
                    /*neg_b=*/false, bfrag[tj].v,
                    /*c_mod=*/(short)0, acc[ti][tj].v,
                    /*reuse_a=*/false, /*reuse_b=*/false);

        __syncthreads();
    }

    // ---- epilogue: out = u * (acc + bias[m]); C/D layout: M=r+8*half, N=lane&15
    #pragma unroll
    for (int ti = 0; ti < 2; ++ti) {
        #pragma unroll
        for (int tj = 0; tj < 2; ++tj) {
            int ncol = n_base + wn * 32 + tj * 16 + ml;
            #pragma unroll
            for (int r = 0; r < 8; ++r) {
                int m = m_base + wm * 32 + ti * 16 + r + 8 * hlf;
                float uval = x[((size_t)b * TT + m) * (2 * HD) + ncol];
                float res  = (acc[ti][tj].f[r] + bias[m]) * uval;
                out[((size_t)b * TT + m) * HD + ncol] = res;
            }
        }
    }
}

// ---------------------------------------------------------------------------
extern "C" void kernel_launch(void* const* d_in, const int* in_sizes, int n_in,
                              void* d_out, int out_size, void* d_ws, size_t ws_size,
                              hipStream_t stream) {
    (void)in_sizes; (void)n_in; (void)out_size; (void)ws_size;
    const float* x     = (const float*)d_in[0];   // [B,T,2H]
    const float* W     = (const float*)d_in[1];   // [T,T]
    const float* bias  = (const float*)d_in[2];   // [T]
    const float* gamma = (const float*)d_in[3];   // [H]
    const float* beta  = (const float*)d_in[4];   // [H]
    float* out = (float*)d_out;                   // [B,T,H]

    bf16* Wbf = (bf16*)d_ws;                                           // T*T bf16 = 2 MB
    bf16* Vn  = (bf16*)((char*)d_ws + (size_t)TT * TT * sizeof(bf16)); // B*T*H bf16 = 32 MB

    // 0) masked cast of W
    sgu_mask_cast_w<<<(TT * TT) / 256, 256, 0, stream>>>(W, Wbf);
    // 1) layernorm of v -> bf16
    sgu_layernorm<<<BSZ * TT, 256, 0, stream>>>(x, gamma, beta, Vn);
    // 2) causal WMMA GEMM + bias + u-gate
    dim3 grid(HD / BN, TT / BM, BSZ);
    sgu_gemm_wmma<<<grid, 256, 0, stream>>>(Wbf, Vn, x, bias, out);
}
